// GraphAttentionLayer_4861902979534
// MI455X (gfx1250) — matrix operand order repaired and checked by reading, not compiled, not run on hardware
//
#include <hip/hip_runtime.h>
#include <hip/hip_bf16.h>

typedef __attribute__((ext_vector_type(16))) __bf16 v16bf;
typedef __attribute__((ext_vector_type(8)))  float  v8f;

#define ALPHA 0.2f
#define LOG2E 1.4426950408889634f

__device__ __forceinline__ float fast_exp(float x) {
    return __builtin_amdgcn_exp2f(x * LOG2E);
}

// ---------------- Pass 1: h = X@W, e_src/e_dst, bf16 pair-packed masked h ----
// Block: 256 threads, 4 rows. h_pack[b][pair][d] : low16 = bf16(h[2p][d]), high16 = bf16(h[2p+1][d])
__global__ void __launch_bounds__(256)
gat_pass1(const float* __restrict__ X, const float* __restrict__ mask,
          const float* __restrict__ W, const float* __restrict__ a,
          unsigned* __restrict__ hpack, float* __restrict__ e_src,
          float* __restrict__ e_dst) {
    __shared__ float Xs[4][128];
    __shared__ float es_part[256];
    __shared__ float ed_part[256];
    const int tid  = threadIdx.x;
    const int row0 = blockIdx.x * 4;            // global row = b*2048 + n
    #pragma unroll
    for (int t = 0; t < 2; ++t) {
        int idx = tid + t * 256;
        ((float*)Xs)[idx] = X[(size_t)row0 * 128 + idx];
    }
    __syncthreads();
    const int r = tid >> 6, d = tid & 63;
    const int row = row0 + r;
    float acc = 0.f;
    #pragma unroll 8
    for (int k = 0; k < 128; ++k)
        acc = fmaf(Xs[r][k], W[k * 64 + d], acc);
    es_part[tid] = acc * a[d];
    ed_part[tid] = acc * a[64 + d];
    const float hm = acc * mask[row];           // mask applied only to h used in einsum
    const int b = row >> 11, n = row & 2047;
    __bf16* hp16 = (__bf16*)hpack;
    hp16[(((size_t)b * 1024 + (n >> 1)) * 64 + d) * 2 + (n & 1)] = (__bf16)hm;
    __syncthreads();
    if (d == 0) {
        float s = 0.f, t2 = 0.f;
        #pragma unroll
        for (int q = 0; q < 64; ++q) { s += es_part[r * 64 + q]; t2 += ed_part[r * 64 + q]; }
        e_src[row] = s;
        e_dst[row] = t2;
    }
}

// ---------------- Pass 2: softmax denominators (one wave per row) ------------
__global__ void __launch_bounds__(256)
gat_pass2(const int* __restrict__ adj, const float* __restrict__ e_src,
          const float* __restrict__ e_dst, float* __restrict__ rdenom) {
    const int tid  = threadIdx.x;
    const int lane = tid & 31;
    const int row  = blockIdx.x * 8 + (tid >> 5);   // b*2048 + i
    const int b    = row >> 11;
    const float esi  = e_src[row];
    const float* edb = e_dst + b * 2048;
    const int*  arow = adj + (size_t)row * 2048;
    float acc = 0.f;
    for (int j0 = 0; j0 < 2048; j0 += 32) {
        const int   j   = j0 + lane;
        const int   ad  = arow[j];
        const float s   = esi + edb[j];
        const float att = ad ? fmaxf(s, ALPHA * s) : -30.0f;  // exp(-30) terms kept, as in ref
        acc += fast_exp(att);
    }
    #pragma unroll
    for (int off = 16; off >= 1; off >>= 1)
        acc += __shfl_xor(acc, off, 32);
    if (lane == 0) rdenom[row] = 1.0f / acc;
}

// ---------------- Pass 3: h' = P^T @ h fused with P generation, ELU ----------
// Block: 128 threads (4 waves), batch b, 64 consecutive j (one 16-j tile/wave).
__global__ void __launch_bounds__(128)
gat_pass3(const int* __restrict__ adj, const unsigned* __restrict__ hpack,
          const float* __restrict__ e_src, const float* __restrict__ e_dst,
          const float* __restrict__ rdenom, float* __restrict__ out) {
    __shared__ float2 g[2048];                       // {e_src[i], rdenom[i]}
    const int tid  = threadIdx.x;
    const int b    = blockIdx.x >> 5;                // 32 blocks per batch
    const int jblk = (blockIdx.x & 31) * 64;
    for (int i = tid; i < 2048; i += 128)
        g[i] = make_float2(e_src[b * 2048 + i], rdenom[b * 2048 + i]);
    __syncthreads();

    const int wave = tid >> 5, lane = tid & 31;
    const int m = lane & 15, H = lane >> 4;
    const int jt = jblk + wave * 16;                 // output j-tile base
    const int j  = jt + m;                           // adj column handled by this lane
    const float edj = e_dst[b * 2048 + j];
    const int*      adjb = adj   + (size_t)b * 2048 * 2048;
    const unsigned* hpb  = hpack + (size_t)b * 1024 * 64;

    v8f acc0 = {}, acc1 = {}, acc2 = {}, acc3 = {};
    union AFrag { v16bf v; __bf16 h[16]; } af;
    union BFrag { v16bf v; unsigned u[8]; } bf0, bf1, bf2, bf3;

    for (int I = 0; I < 2048; I += 32) {
        // Build A fragment = P^T tile (16 j x 32 i) in contiguous k order:
        // VGPR v, lane-half H: K = 2*v + 16*H  (+1 for high half)
        #pragma unroll
        for (int v = 0; v < 8; ++v) {
            const int kb = (v << 1) + (H << 4);
            const int i0 = I + kb;
            const int a0 = adjb[(size_t)i0 * 2048 + j];           // coalesced 64B/half-wave
            const int a1 = adjb[(size_t)(i0 + 1) * 2048 + j];
            const float2 g0 = g[i0], g1 = g[i0 + 1];              // LDS broadcast
            const float s0 = g0.x + edj, s1 = g1.x + edj;
            const float att0 = a0 ? fmaxf(s0, ALPHA * s0) : -30.0f;
            const float att1 = a1 ? fmaxf(s1, ALPHA * s1) : -30.0f;
            const float p0 = fast_exp(att0) * g0.y;
            const float p1 = fast_exp(att1) * g1.y;
            af.h[2 * v]     = (__bf16)p0;
            af.h[2 * v + 1] = (__bf16)p1;
        }
        // B fragments from pair-packed h: VGPR v = pair (I/2 + v + 8H), col = d
        const unsigned* hrow = hpb + (size_t)((I >> 1) + (H << 3)) * 64 + m;
        #pragma unroll
        for (int v = 0; v < 8; ++v) {
            bf0.u[v] = hrow[(size_t)v * 64 +  0];
            bf1.u[v] = hrow[(size_t)v * 64 + 16];
            bf2.u[v] = hrow[(size_t)v * 64 + 32];
            bf3.u[v] = hrow[(size_t)v * 64 + 48];
        }
        acc0 = __builtin_amdgcn_wmma_f32_16x16x32_bf16(false, af.v, false, bf0.v, (short)0, acc0, false, false);
        acc1 = __builtin_amdgcn_wmma_f32_16x16x32_bf16(false, af.v, false, bf1.v, (short)0, acc1, false, false);
        acc2 = __builtin_amdgcn_wmma_f32_16x16x32_bf16(false, af.v, false, bf2.v, (short)0, acc2, false, false);
        acc3 = __builtin_amdgcn_wmma_f32_16x16x32_bf16(false, af.v, false, bf3.v, (short)0, acc3, false, false);
        if (I + 32 < 2048)
            __builtin_prefetch(adjb + (size_t)(I + 32) * 2048 + j, 0, 0);
    }

    // ELU + store. D layout: lane L, VGPR r -> row M=r+8H, col N=L%16
    float* ob = out + (size_t)b * 2048 * 64;
    #pragma unroll
    for (int r = 0; r < 8; ++r) {
        const int jo = jt + r + (H << 3);
        float x;
        x = acc0[r]; ob[(size_t)jo * 64 + m +  0] = x > 0.f ? x : fast_exp(x) - 1.f;
        x = acc1[r]; ob[(size_t)jo * 64 + m + 16] = x > 0.f ? x : fast_exp(x) - 1.f;
        x = acc2[r]; ob[(size_t)jo * 64 + m + 32] = x > 0.f ? x : fast_exp(x) - 1.f;
        x = acc3[r]; ob[(size_t)jo * 64 + m + 48] = x > 0.f ? x : fast_exp(x) - 1.f;
    }
}

// ---------------- launcher ---------------------------------------------------
extern "C" void kernel_launch(void* const* d_in, const int* in_sizes, int n_in,
                              void* d_out, int out_size, void* d_ws, size_t ws_size,
                              hipStream_t stream) {
    (void)in_sizes; (void)n_in; (void)out_size; (void)ws_size;
    const float* X    = (const float*)d_in[0];   // [8,2048,128]
    const int*   adj  = (const int*)  d_in[1];   // [8,2048,2048]
    const float* mask = (const float*)d_in[2];   // [8,2048]
    const float* W    = (const float*)d_in[3];   // [128,64]
    const float* a    = (const float*)d_in[4];   // [128,1]
    float* out = (float*)d_out;                  // [8,2048,64]

    char* ws = (char*)d_ws;
    unsigned* hpack  = (unsigned*)ws;                      // 8*1024*64*4 = 2 MB
    float*    e_src  = (float*)(ws + (size_t)8 * 1024 * 64 * 4);
    float*    e_dst  = e_src + 16384;
    float*    rdenom = e_dst + 16384;

    gat_pass1<<<4096, 256, 0, stream>>>(X, mask, W, a, hpack, e_src, e_dst);
    gat_pass2<<<2048, 256, 0, stream>>>(adj, e_src, e_dst, rdenom);
    gat_pass3<<<256, 128, 0, stream>>>(adj, hpack, e_src, e_dst, rdenom, out);
}
